// GCNLayer_4569845203241
// MI455X (gfx1250) — matrix operand order repaired and checked by reading, not compiled, not run on hardware
//
#include <hip/hip_runtime.h>

typedef __attribute__((ext_vector_type(16))) _Float16 v16h;
typedef __attribute__((ext_vector_type(8)))  _Float16 v8h;
typedef __attribute__((ext_vector_type(4)))  _Float16 v4h;
typedef __attribute__((ext_vector_type(8)))  float    v8f;
typedef __attribute__((ext_vector_type(4)))  float    v4f;

#define NN 8192   // number of nodes
#define CD 128    // channels (C_IN == C_OUT)
#define KC 64     // K-chunk for stage 2 (2 WMMA K-steps)
#define LDP 72    // padded LDS row stride (halfs): 144B -> conflict-free b128 reads

// Convert 8 contiguous fp32 values to 8 packed f16 (loads as 2x dwordx4).
static __device__ inline v8h cvt8(const float* __restrict__ p) {
    v4f a = *(const v4f*)p;
    v4f b = *(const v4f*)(p + 4);
    v8h r;
#pragma unroll
    for (int i = 0; i < 4; ++i) { r[i] = (_Float16)a[i]; r[i + 4] = (_Float16)b[i]; }
    return r;
}

static __device__ inline v16h cat16(v8h lo, v8h hi) {
    return __builtin_shufflevector(lo, hi, 0,1,2,3,4,5,6,7,8,9,10,11,12,13,14,15);
}

// ---------------------------------------------------------------------------
// Stage 1: h_t = (x @ W^T)^T stored as f16, shape [CD][NN] (row = out channel).
// A = W (row-major, contiguous in K=c); B = x^T so a B-column is a contiguous
// x row. 4096 waves = 8 M-tiles x 512 N-tiles; 512 blocks x 8 waves.
// ---------------------------------------------------------------------------
__global__ __launch_bounds__(256) void gcn_stage1(
    const float* __restrict__ x,   // [NN][CD] fp32
    const float* __restrict__ W,   // [CD][CD] fp32
    _Float16* __restrict__ ht)     // [CD][NN] f16 (transposed h)
{
    const int lane = threadIdx.x & 31;
    const int wave = threadIdx.x >> 5;
    const int gw   = blockIdx.x * 8 + wave;
    const int mt   = gw & 7;        // tile over out-channels
    const int nt   = gw >> 3;       // tile over nodes
    const int half = lane >> 4;
    const int lq   = lane & 15;

    const float* wrow = W + (mt * 16 + lq) * CD;          // A-matrix row (lane m)
    const float* xrow = x + (long)(nt * 16 + lq) * CD;    // B-matrix col = x row

    v8f acc = {};
#pragma unroll
    for (int k0 = 0; k0 < CD; k0 += 32) {
        v8h a_lo = cvt8(wrow + k0 + half * 8);
        v8h a_hi = cvt8(wrow + k0 + 16 + half * 8);
        v8h b_lo = cvt8(xrow + k0 + half * 8);
        v8h b_hi = cvt8(xrow + k0 + 16 + half * 8);
        acc = __builtin_amdgcn_wmma_f32_16x16x32_f16(
            false, cat16(a_lo, a_hi), false, cat16(b_lo, b_hi),
            (short)0, acc, false, false);
    }
#pragma unroll
    for (int r = 0; r < 8; ++r) {
        int orow = mt * 16 + half * 8 + r;
        ht[(long)orow * NN + nt * 16 + lq] = (_Float16)acc[r];
    }
}

// ---------------------------------------------------------------------------
// Stage 2: out = (adj*mask) @ h + h   (identity folded out of the hot loop).
// Block = 16-row strip; 8 waves each own one 16-col tile. adj/mask staged once
// per block into LDS as f16 (NT global b128 loads, prefetched one chunk ahead,
// LDS double-buffered, 1 barrier per 64-K chunk = per 2 WMMAs).
// ---------------------------------------------------------------------------
__global__ __launch_bounds__(256) void gcn_stage2(
    const float* __restrict__ adj,    // [NN][NN] fp32
    const float* __restrict__ mask,   // [NN][NN] fp32
    const _Float16* __restrict__ ht,  // [CD][NN] f16
    float* __restrict__ out)          // [NN][CD] fp32
{
    __shared__ _Float16 a_lds[2][16 * LDP];

    const int tid  = threadIdx.x;
    const int lane = tid & 31;
    const int wave = tid >> 5;
    const int half = lane >> 4;
    const int lq   = lane & 15;
    const int r0   = blockIdx.x * 16;   // output row base for this block
    const int c0   = wave * 16;         // output col base for this wave

    // Cooperative staging: 16x64 fp32 elems per chunk, 4 per thread (float4).
    const int  j    = tid * 4;
    const int  srow = j >> 6;           // 0..15
    const int  scol = j & 63;           // multiple of 4
    const float* ap = adj  + (long)(r0 + srow) * NN + scol;
    const float* mp = mask + (long)(r0 + srow) * NN + scol;

    // B fragment base: lane reads ht row (c0+lq), contiguous in K.
    const _Float16* brow = ht + (long)(c0 + lq) * NN;

    // Prefetch first chunk (streaming data -> non-temporal).
    v4f anx = __builtin_nontemporal_load((const v4f*)ap);
    v4f mnx = __builtin_nontemporal_load((const v4f*)mp);

    v8f acc = {};
    for (int k0 = 0; k0 < NN; k0 += KC) {
        const int buf = (k0 >> 6) & 1;
        v4f acur = anx, mcur = mnx;
        // Prefetch next chunk (wraps harmlessly on the last iteration).
        const int kn = (k0 + KC) & (NN - 1);
        anx = __builtin_nontemporal_load((const v4f*)(ap + kn));
        mnx = __builtin_nontemporal_load((const v4f*)(mp + kn));

        // a_eff = adj*mask, f32 -> f16, into padded LDS
        v4h pk;
#pragma unroll
        for (int i = 0; i < 4; ++i) pk[i] = (_Float16)(acur[i] * mcur[i]);
        *(v4h*)&a_lds[buf][srow * LDP + scol] = pk;
        __syncthreads();

        const _Float16* arow = &a_lds[buf][lq * LDP];
#pragma unroll
        for (int sub = 0; sub < 2; ++sub) {
            const int kb = k0 + sub * 32;
            v8h a_lo = *(const v8h*)(arow + sub * 32 + half * 8);
            v8h a_hi = *(const v8h*)(arow + sub * 32 + 16 + half * 8);
            v8h b_lo = *(const v8h*)(brow + kb + half * 8);
            v8h b_hi = *(const v8h*)(brow + kb + 16 + half * 8);
            acc = __builtin_amdgcn_wmma_f32_16x16x32_f16(
                false, cat16(a_lo, a_hi), false, cat16(b_lo, b_hi),
                (short)0, acc, false, false);
        }
    }

    // out = acc + h  (identity contribution), h read from transposed f16 h_t.
#pragma unroll
    for (int r = 0; r < 8; ++r) {
        int row = r0 + half * 8 + r;
        out[(long)row * CD + c0 + lq] = acc[r] + (float)brow[row];
    }
}

extern "C" void kernel_launch(void* const* d_in, const int* in_sizes, int n_in,
                              void* d_out, int out_size, void* d_ws, size_t ws_size,
                              hipStream_t stream) {
    (void)in_sizes; (void)n_in; (void)out_size; (void)ws_size;
    const float* x    = (const float*)d_in[0];
    const float* adj  = (const float*)d_in[1];
    const float* mask = (const float*)d_in[2];
    const float* W    = (const float*)d_in[3];
    float* out = (float*)d_out;
    _Float16* ht = (_Float16*)d_ws;   // 128*8192 f16 = 2 MB scratch

    gcn_stage1<<<512, 256, 0, stream>>>(x, W, ht);
    gcn_stage2<<<512, 256, 0, stream>>>(adj, mask, ht, out);
}
